// AnyprecisionLinear_25417616457968
// MI455X (gfx1250) — compile-verified, matrix operand order checked by reading, not run on hardware
//
#include <hip/hip_runtime.h>

typedef __attribute__((ext_vector_type(16))) _Float16 v16h;
typedef __attribute__((ext_vector_type(8)))  _Float16 v8h;
typedef __attribute__((ext_vector_type(8)))  float    v8f;
typedef __attribute__((ext_vector_type(4)))  unsigned int v4u;

#define BATCH    64
#define IN_DIM   8192
#define OUT_DIM  8192
#define NROWS    64     // weight rows (N) per block
#define THREADS  128    // 4 waves of 32

union AB16 {
    v16h h16;
    v8h  h8[2];
    unsigned u32[8];
};

__global__ __launch_bounds__(THREADS)
void anyprec_lut4_gemm(const _Float16* __restrict__ x,
                       const int*      __restrict__ qw,
                       const unsigned short* __restrict__ lut, // fp16 bits
                       float*          __restrict__ out)
{
    // 64 rows x 256 packed-fp16-pair entries = 64 KB
    __shared__ unsigned lutp[NROWS * 256];

    const int tid  = threadIdx.x;
    const int lane = tid & 31;
    const int wave = tid >> 5;                 // 0..3 -> N sub-tile
    const int rowBase = blockIdx.x * NROWS;    // first weight row of block

    // ---- Build pair LUT: entry p packs lut[p&15] (lo half) | lut[p>>4] (hi half)
    for (int e = tid; e < NROWS * 256; e += THREADS) {
        const int r = e >> 8;
        const int p = e & 255;
        const unsigned short* lrow = lut + (size_t)(rowBase + r) * 16;
        const unsigned lo = lrow[p & 15];
        const unsigned hi = lrow[p >> 4];
        lutp[e] = lo | (hi << 16);
    }
    __syncthreads();

    // ---- Per-lane geometry (16-bit WMMA operand striping)
    const int nl = lane & 15;                  // N within 16-wide tile / M row within A tile
    const int kh = (lane >> 4) << 3;           // K sub-offset: 0 (lanes 0-15) or 8 (lanes 16-31)
    const int nLocal = wave * 16 + nl;         // 0..63
    const int nGlob  = rowBase + nLocal;       // absolute weight row

    const int* qrow = qw + (size_t)nGlob * IN_DIM;
    // Byte base of this lane's pair-LUT row; per-lookup addr = lbase + (q0<<2) + (q1<<6)
    const char* lbase = (const char*)lutp + nLocal * 1024;

    v8f acc0 = {}, acc1 = {}, acc2 = {}, acc3 = {};

    for (int kk = 0; kk < IN_DIM; kk += 32) {
        // Prefetch the index stream ~8 iterations ahead (global_prefetch_b8)
        __builtin_prefetch(qrow + kk + 256, 0, 1);

        // qweight: two contiguous runs of 8 int32 (k = kk+kh..+7 and kk+16+kh..+7)
        const v4u qa = *(const v4u*)(qrow + kk + kh);
        const v4u qb = *(const v4u*)(qrow + kk + kh + 4);
        const v4u qc = *(const v4u*)(qrow + kk + kh + 16);
        const v4u qd = *(const v4u*)(qrow + kk + kh + 20);

        // Dequant: each B dword = two consecutive K elements via one LDS pair lookup.
        // qweight values are guaranteed 0..15, so no masking: addr = lbase + q0*4 + q1*64.
        AB16 b;
        b.u32[0] = *(const unsigned*)(lbase + (qa[0] << 2) + (qa[1] << 6));
        b.u32[1] = *(const unsigned*)(lbase + (qa[2] << 2) + (qa[3] << 6));
        b.u32[2] = *(const unsigned*)(lbase + (qb[0] << 2) + (qb[1] << 6));
        b.u32[3] = *(const unsigned*)(lbase + (qb[2] << 2) + (qb[3] << 6));
        b.u32[4] = *(const unsigned*)(lbase + (qc[0] << 2) + (qc[1] << 6));
        b.u32[5] = *(const unsigned*)(lbase + (qc[2] << 2) + (qc[3] << 6));
        b.u32[6] = *(const unsigned*)(lbase + (qd[0] << 2) + (qd[1] << 6));
        b.u32[7] = *(const unsigned*)(lbase + (qd[2] << 2) + (qd[3] << 6));

        // 4 M-tiles share this B tile: 4 x v_wmma_f32_16x16x32_f16
        #pragma unroll
        for (int m = 0; m < 4; ++m) {
            AB16 a;
            const _Float16* xrow = x + (size_t)(m * 16 + nl) * IN_DIM + kk + kh;
            a.h8[0] = *(const v8h*)(xrow);
            a.h8[1] = *(const v8h*)(xrow + 16);

            v8f* accp = (m == 0) ? &acc0 : (m == 1) ? &acc1 : (m == 2) ? &acc2 : &acc3;
            *accp = __builtin_amdgcn_wmma_f32_16x16x32_f16(
                false, a.h16, false, b.h16, (short)0, *accp, false, false);
        }
    }

    // ---- Store C tiles: lane holds (M = mt*16 + v + 8*(lane>=16), N = nGlob), f32 out
    const int mhi = (lane >> 4) << 3;
    #pragma unroll
    for (int v = 0; v < 8; ++v) {
        out[(size_t)( 0 + v + mhi) * OUT_DIM + nGlob] = acc0[v];
        out[(size_t)(16 + v + mhi) * OUT_DIM + nGlob] = acc1[v];
        out[(size_t)(32 + v + mhi) * OUT_DIM + nGlob] = acc2[v];
        out[(size_t)(48 + v + mhi) * OUT_DIM + nGlob] = acc3[v];
    }
}

extern "C" void kernel_launch(void* const* d_in, const int* in_sizes, int n_in,
                              void* d_out, int out_size, void* d_ws, size_t ws_size,
                              hipStream_t stream) {
    const _Float16*       x   = (const _Float16*)d_in[0];
    const int*            qw  = (const int*)d_in[1];
    const unsigned short* lut = (const unsigned short*)d_in[2];
    float*                out = (float*)d_out;

    dim3 grid(OUT_DIM / NROWS);   // 128 blocks
    dim3 block(THREADS);          // 4 waves
    anyprec_lut4_gemm<<<grid, block, 0, stream>>>(x, qw, lut, out);
}
